// SnailBlock_20804821582359
// MI455X (gfx1250) — compile-verified, matrix operand order checked by reading, not run on hardware
//
#include <hip/hip_runtime.h>
#include <hip/hip_bf16.h>
#include <stdint.h>
#include <stddef.h>

#define DI __device__ __forceinline__

typedef __attribute__((ext_vector_type(16))) __bf16 v16bf;
typedef __attribute__((ext_vector_type(8)))  float  v8f;

union ABFrag { v16bf v; unsigned u[8]; };

#define MPIX 16384   // B*H*W = 4*64*64
#define NPOS 4096    // H*W
#define HID  128

// ---- CDNA5 async copy to LDS (ASYNCcnt) -----------------------------------
DI void async_copy16(unsigned lds_off, const void* gaddr) {
  // GLOBAL_LOAD_ASYNC_TO_LDS_B128, GV mode: VDST = LDS byte address VGPR
  asm volatile("global_load_async_to_lds_b128 %0, %1, off"
               :: "v"(lds_off), "v"(gaddr) : "memory");
}
#define WAIT_ASYNC0()  asm volatile("s_wait_asynccnt 0x0" ::: "memory")
#define WAIT_ASYNC16() asm volatile("s_wait_asynccnt 0x10" ::: "memory")
#define WAIT_DS0()     asm volatile("s_wait_dscnt 0x0" ::: "memory")

// K-pair base index for 16-bit WMMA A/B fragments (wave32):
// element pair j (0..7) of lane (m|n = lane&15, half = lane>>4)
DI int kidx(int j, int half) {
  return ((j & 3) << 1) + (half << 3) + ((j >> 2) << 4);
}

DI v8f bwmma(v16bf a, v16bf b, v8f c) {
  return __builtin_amdgcn_wmma_f32_16x16x32_bf16(false, a, false, b, (short)0, c,
                                                 false, false);
}

DI unsigned short f2bf(float f) {
  unsigned u = __float_as_uint(f);
  u += 0x7FFFu + ((u >> 16) & 1u);   // round-to-nearest-even
  return (unsigned short)(u >> 16);
}

DI float eluf(float x) { return x > 0.f ? x : (__expf(x) - 1.f); }
DI float sigf(float x) { return 1.f / (1.f + __expf(-x)); }

// ---------------------------------------------------------------------------
// Elementwise / repack kernels
// ---------------------------------------------------------------------------

__global__ void transpose_in_kernel(const float* __restrict__ x, float* __restrict__ h) {
  int idx = blockIdx.x * blockDim.x + threadIdx.x;
  if (idx >= MPIX * HID) return;
  int m = idx >> 7, c = idx & 127;
  int b = m >> 12, p = m & 4095;
  h[idx] = x[((size_t)(b * HID + c)) * NPOS + p];
}

__global__ void elu_bf16_kernel(const float* __restrict__ s, unsigned short* __restrict__ d, int n) {
  int idx = blockIdx.x * blockDim.x + threadIdx.x;
  if (idx >= n) return;
  d[idx] = f2bf(eluf(s[idx]));
}

__global__ void add_elu_bf16_kernel(const float* __restrict__ a, const float* __restrict__ b,
                                    unsigned short* __restrict__ d, int n) {
  int idx = blockIdx.x * blockDim.x + threadIdx.x;
  if (idx >= n) return;
  d[idx] = f2bf(eluf(a[idx] + b[idx]));
}

// h[m][c] = elu(h[m][c]) + o[m][c] * sigmoid(o[m][128+c])   (in-place, elementwise)
__global__ void gate_kernel(float* __restrict__ h, const float* __restrict__ o) {
  int idx = blockIdx.x * blockDim.x + threadIdx.x;
  if (idx >= MPIX * HID) return;
  int m = idx >> 7, c = idx & 127;
  float o1 = o[(size_t)m * 256 + c];
  float o2 = o[(size_t)m * 256 + 128 + c];
  h[idx] = eluf(h[idx]) + o1 * sigf(o2);
}

// abuf[m][0..127]=bf16(h), [128..129]=bf16(pos), [130..159]=0  (K padded to 160)
__global__ void build_a_kernel(const float* __restrict__ h, const float* __restrict__ pos,
                               unsigned short* __restrict__ abuf) {
  int idx = blockIdx.x * blockDim.x + threadIdx.x;
  if (idx >= MPIX * 160) return;
  int m = idx / 160, j = idx % 160;
  int b = m >> 12, p = m & 4095;
  float v = 0.f;
  if (j < 128)      v = h[(size_t)m * HID + j];
  else if (j < 130) v = pos[((size_t)(b * 2 + (j - 128))) * NPOS + p];
  abuf[idx] = f2bf(v);
}

// split qkv f32 [M][160] -> qb[M][16], kb[M][16], vtb[B][128][N] (all bf16)
__global__ void pack_qkv_kernel(const float* __restrict__ qkv,
                                unsigned short* __restrict__ qb,
                                unsigned short* __restrict__ kb,
                                unsigned short* __restrict__ vtb) {
  int idx = blockIdx.x * blockDim.x + threadIdx.x;
  if (idx >= MPIX * 160) return;
  int m = idx / 160, j = idx % 160;
  unsigned short v = f2bf(qkv[idx]);
  if (j < 16)       qb[(size_t)m * 16 + j] = v;
  else if (j < 32)  kb[(size_t)m * 16 + (j - 16)] = v;
  else {
    int c = j - 32, b = m >> 12, p = m & 4095;
    vtb[((size_t)(b * HID + c)) * NPOS + p] = v;
  }
}

// conv weights [L][O][128][2][2] f32 -> [L][O][512] bf16, K = tap*128 + c
__global__ void prep_conv_w_kernel(const float* __restrict__ w, unsigned short* __restrict__ dst,
                                   int total) {
  int idx = blockIdx.x * blockDim.x + threadIdx.x;
  if (idx >= total) return;
  int k = idx & 511, lo = idx >> 9;
  int tap = k >> 7, c = k & 127;
  dst[idx] = f2bf(w[((size_t)(lo * 128 + c)) * 4 + tap]);
}

// linear weights [O][C] f32 -> [O][KP] bf16 (zero padded)
__global__ void prep_lin_w_kernel(const float* __restrict__ w, unsigned short* __restrict__ dst,
                                  int O, int C, int KP) {
  int idx = blockIdx.x * blockDim.x + threadIdx.x;
  if (idx >= O * KP) return;
  int o = idx / KP, k = idx % KP;
  dst[idx] = f2bf(k < C ? w[(size_t)o * C + k] : 0.f);
}

// ---------------------------------------------------------------------------
// Implicit-GEMM 2x2 causal conv. 4 waves/WG, each wave: 16 pixels x 128 cols.
// Weight slab (128 cols x 32 K = 8 KB) staged per K-step into LDS via
// global_load_async_to_lds_b128, double buffered.
// ---------------------------------------------------------------------------
__global__ __launch_bounds__(128) void conv_gemm_kernel(
    const unsigned short* __restrict__ inb,   // bf16 [M][128] pixel-major
    const unsigned short* __restrict__ Bw,    // bf16 [Oc][512]
    const float* __restrict__ bias,           // [Oc]
    float* __restrict__ Of,                   // optional f32 [M][Oc]
    unsigned short* __restrict__ Ob,          // optional bf16 [M][Oc]
    int Oc, int act) {
  __shared__ __align__(16) unsigned short wlds[2][128 * 32];
  int tid = threadIdx.x;
  int wave = tid >> 5, lane = tid & 31;
  int n = lane & 15, half = lane >> 4;
  int m0 = (blockIdx.x * 4 + wave) * 16;       // 16 consecutive pixels, same row
  int n0base = blockIdx.y * 128;
  int b = m0 >> 12, y = (m0 >> 6) & 63, x0 = m0 & 63;
  int x = x0 + n;

  // cooperative async stage of the 8 KB weight slab for K-step ks
  auto stage = [&](int ks, int buf) {
    int ci = tid * 4;                       // chunks 0..511 (16 B each)
    for (int q = 0; q < 4; ++q, ++ci) {
      int col = ci >> 2, part = ci & 3;
      const char* g = (const char*)(Bw + (size_t)(n0base + col) * 512 + ks * 32) + part * 16;
      unsigned l = (unsigned)(size_t)&wlds[buf][col * 32] + part * 16;
      async_copy16(l, g);
    }
  };

  v8f acc[8];
  for (int nt = 0; nt < 8; ++nt) {
    float bv = bias[n0base + nt * 16 + n];
    for (int r = 0; r < 8; ++r) acc[nt][r] = bv;
  }

  stage(0, 0);
  WAIT_ASYNC0();
  __syncthreads();

  for (int ks = 0; ks < 16; ++ks) {
    int cur = ks & 1;
    if (ks < 15) stage(ks + 1, cur ^ 1);    // prefetch next slab (overlapped)

    int tap = ks >> 2;                      // 4 K-steps per tap
    int yy = y - 1 + (tap >> 1);
    int xx = x - 1 + (tap & 1);
    bool ok = (yy >= 0) && (xx >= 0);
    const unsigned short* src = inb + ((size_t)(b * NPOS + yy * 64 + xx)) * HID;
    ABFrag af;
    for (int j = 0; j < 8; ++j) {
      int c = (ks & 3) * 32 + kidx(j, half);
      af.u[j] = ok ? *(const unsigned*)(src + c) : 0u;
    }
    for (int nt = 0; nt < 8; ++nt) {
      const unsigned short* brow = &wlds[cur][(nt * 16 + n) * 32];
      ABFrag bf;
      for (int j = 0; j < 8; ++j) bf.u[j] = *(const unsigned*)(brow + kidx(j, half));
      acc[nt] = bwmma(af.v, bf.v, acc[nt]);
    }
    if (ks < 15) {
      WAIT_ASYNC0();                        // next slab landed
      __syncthreads();                      // all waves done reading cur
    }
  }

  for (int nt = 0; nt < 8; ++nt)
    for (int r = 0; r < 8; ++r) {
      float v = acc[nt][r];
      if (act) v = eluf(v);
      int m = m0 + r + 8 * half;
      int col = n0base + nt * 16 + n;
      if (Of) Of[(size_t)m * Oc + col] = v;
      if (Ob) Ob[(size_t)m * Oc + col] = f2bf(v);
    }
}

// ---------------------------------------------------------------------------
// Generic 1x1-conv GEMM. Whole weight matrix (<= 50 KB) async-staged into LDS
// once per workgroup; 4 waves consume it.
// ---------------------------------------------------------------------------
__global__ __launch_bounds__(128) void gemm1x1_kernel(
    const unsigned short* __restrict__ A,   // bf16 [M][KP]
    const unsigned short* __restrict__ Bw,  // bf16 [Oc][KP]
    const float* __restrict__ bias,
    float* __restrict__ Cf,                 // optional f32 [M][Oc]
    unsigned short* __restrict__ Cb,        // optional bf16 [M][Oc]
    float* __restrict__ Cn,                 // optional f32 NCHW [B][Oc][H][W]
    int KP, int ntiles, int Oc, int act) {
  __shared__ __align__(16) unsigned short wlds[160 * 160];
  int tid = threadIdx.x;
  int wave = tid >> 5, lane = tid & 31;
  int n = lane & 15, half = lane >> 4;
  int m0 = (blockIdx.x * 4 + wave) * 16;
  int ksteps = KP >> 5;

  // async stage the full [Oc][KP] weight matrix
  int nchunks = (Oc * KP) >> 3;             // 16-byte chunks
  for (int ci = tid; ci < nchunks; ci += 128)
    async_copy16((unsigned)(size_t)wlds + ci * 16, (const char*)Bw + ci * 16);

  // overlap: preload A fragments from global while weights stream in
  ABFrag a[5];
  const unsigned short* arow = A + (size_t)(m0 + n) * KP;
  for (int ks = 0; ks < ksteps; ++ks)
    for (int j = 0; j < 8; ++j)
      a[ks].u[j] = *(const unsigned*)(arow + ks * 32 + kidx(j, half));

  WAIT_ASYNC0();
  __syncthreads();

  for (int nt = 0; nt < ntiles; ++nt) {
    int col = nt * 16 + n;
    float bv = bias[col];
    v8f acc;
    for (int r = 0; r < 8; ++r) acc[r] = bv;
    const unsigned short* brow = &wlds[(size_t)col * KP];
    for (int ks = 0; ks < ksteps; ++ks) {
      ABFrag bf;
      for (int j = 0; j < 8; ++j) bf.u[j] = *(const unsigned*)(brow + ks * 32 + kidx(j, half));
      acc = bwmma(a[ks].v, bf.v, acc);
    }
    for (int r = 0; r < 8; ++r) {
      float v = acc[r];
      if (act) v = eluf(v);
      int m = m0 + r + 8 * half;
      if (Cf) Cf[(size_t)m * Oc + col] = v;
      if (Cb) Cb[(size_t)m * Oc + col] = f2bf(v);
      if (Cn) {
        int bb = m >> 12, p = m & 4095;
        Cn[((size_t)(bb * Oc + col)) * NPOS + p] = v;
      }
    }
  }
}

// ---------------------------------------------------------------------------
// Causal flash attention. One wave per 16-query tile. V tiles (8 KB / 32 keys)
// double-buffered in LDS via async copies with a counted s_wait_asynccnt so the
// next tile's prefetch stays in flight during the current P.V WMMAs.
// ---------------------------------------------------------------------------
__global__ __launch_bounds__(32) void attn_kernel(
    const unsigned short* __restrict__ qb,   // bf16 [B*N][16]
    const unsigned short* __restrict__ kb,   // bf16 [B*N][16]
    const unsigned short* __restrict__ vtb,  // bf16 [B][128][N]
    unsigned short* __restrict__ avb) {      // bf16 [B*N][128]
  int lane = threadIdx.x;
  int tile = blockIdx.x;          // b*256 + qtile
  int b = tile >> 8, qt = tile & 255;
  int q0 = qt * 16;
  int n = lane & 15, half = lane >> 4;

  __shared__ __align__(16) unsigned short plds[16 * 32];
  __shared__ __align__(16) unsigned short vlds[2][128 * 32];

  // stage a 32-key V tile (128 rows x 64 B) into vlds[buf]: 16 async ops/lane
  auto stageV = [&](int j0, int buf) {
    for (int q = 0; q < 16; ++q) {
      int ci = q * 32 + lane;               // 512 chunks of 16 B
      int c = ci >> 2, part = ci & 3;
      const char* g = (const char*)(vtb + ((size_t)(b * HID + c)) * NPOS + j0) + part * 16;
      async_copy16((unsigned)(size_t)&vlds[buf][c * 32] + part * 16, g);
    }
  };

  ABFrag qf;
  {
    const unsigned short* qrow = qb + (size_t)(b * NPOS + q0 + n) * 16;
    for (int j = 0; j < 8; ++j) {
      int kk = kidx(j, half);
      qf.u[j] = (kk < 16) ? *(const unsigned*)(qrow + kk) : 0u;
    }
  }

  float rowmax[8], rowsum[8];
  v8f acc[8];
  for (int r = 0; r < 8; ++r) { rowmax[r] = -__builtin_inff(); rowsum[r] = 0.f; }
  for (int t = 0; t < 8; ++t)
    for (int r = 0; r < 8; ++r) acc[t][r] = 0.f;

  stageV(0, 0);

  int it = 0;
  for (int j0 = 0; j0 <= q0 + 15; j0 += 32, ++it) {
    int cur = it & 1;
    bool hasNext = (j0 + 32) <= (q0 + 15);
    if (hasNext) {
      WAIT_DS0();                 // async LDS writes are unordered vs ds reads
      stageV(j0 + 32, cur ^ 1);   // prefetch next V tile
    }

    // ---- scores: S = q . K^T (K fragments straight from global / L2)
    ABFrag bf0, bf1;
    const unsigned short* kbase = kb + (size_t)(b * NPOS + j0) * 16;
    for (int j = 0; j < 8; ++j) {
      int kk = kidx(j, half);
      bf0.u[j] = (kk < 16) ? *(const unsigned*)(kbase + (size_t)n * 16 + kk) : 0u;
      bf1.u[j] = (kk < 16) ? *(const unsigned*)(kbase + (size_t)(16 + n) * 16 + kk) : 0u;
    }
    v8f z;
    for (int r = 0; r < 8; ++r) z[r] = 0.f;
    v8f s0 = bwmma(qf.v, bf0.v, z);
    v8f s1 = bwmma(qf.v, bf1.v, z);

    // ---- scale + causal mask + online softmax
    float sf[8];
    for (int r = 0; r < 8; ++r) {
      int g = q0 + r + 8 * half;
      float v0 = s0[r] * 0.25f;
      float v1 = s1[r] * 0.25f;
      if (j0 + n > g)      v0 = -__builtin_inff();
      if (j0 + 16 + n > g) v1 = -__builtin_inff();
      float mx = fmaxf(v0, v1);
      for (int off = 8; off >= 1; off >>= 1) mx = fmaxf(mx, __shfl_xor(mx, off, 32));
      float nm = fmaxf(rowmax[r], mx);
      sf[r] = __expf(rowmax[r] - nm);
      rowmax[r] = nm;
      float p0 = __expf(v0 - nm);
      float p1 = __expf(v1 - nm);
      s0[r] = p0; s1[r] = p1;
      float rs = p0 + p1;
      for (int off = 8; off >= 1; off >>= 1) rs += __shfl_xor(rs, off, 32);
      rowsum[r] = rowsum[r] * sf[r] + rs;
    }
    for (int t = 0; t < 8; ++t)
      for (int r = 0; r < 8; ++r) acc[t][r] *= sf[r];

    // ---- transpose P (C layout) into an A fragment via LDS
    for (int r = 0; r < 8; ++r) {
      int m = r + 8 * half;
      plds[m * 32 + n]      = f2bf(s0[r]);
      plds[m * 32 + 16 + n] = f2bf(s1[r]);
    }
    __syncthreads();
    ABFrag pf;
    for (int j = 0; j < 8; ++j)
      pf.u[j] = *(const unsigned*)(plds + (size_t)n * 32 + kidx(j, half));
    __syncthreads();

    // current V tile ready; next tile's 16 copies may stay outstanding
    if (hasNext) { WAIT_ASYNC16(); } else { WAIT_ASYNC0(); }

    // ---- O += P . V  from LDS (8 x 16-col V subtiles, K = 32 keys)
    for (int t = 0; t < 8; ++t) {
      const unsigned short* vrow = &vlds[cur][(t * 16 + n) * 32];
      ABFrag vf;
      for (int j = 0; j < 8; ++j) vf.u[j] = *(const unsigned*)(vrow + kidx(j, half));
      acc[t] = bwmma(pf.v, vf.v, acc[t]);
    }
  }

  for (int t = 0; t < 8; ++t)
    for (int r = 0; r < 8; ++r) {
      float o = acc[t][r] / rowsum[r];
      int m = q0 + r + 8 * half;
      avb[(size_t)(b * NPOS + m) * HID + t * 16 + n] = f2bf(o);
    }
}

// ---------------------------------------------------------------------------
// Host launcher
// ---------------------------------------------------------------------------
static inline int cdiv(int a, int b) { return (a + b - 1) / b; }

extern "C" void kernel_launch(void* const* d_in, const int* in_sizes, int n_in,
                              void* d_out, int out_size, void* d_ws, size_t ws_size,
                              hipStream_t stream) {
  (void)in_sizes; (void)n_in; (void)out_size; (void)ws_size;

  const float* x          = (const float*)d_in[0];
  const float* pos        = (const float*)d_in[1];
  const float* conv1_w    = (const float*)d_in[2];
  const float* conv1_b    = (const float*)d_in[3];
  const float* conv2_w    = (const float*)d_in[4];
  const float* conv2_b    = (const float*)d_in[5];
  const float* qkv_w      = (const float*)d_in[6];
  const float* qkv_b      = (const float*)d_in[7];
  const float* ap_w       = (const float*)d_in[8];
  const float* ap_b       = (const float*)d_in[9];
  const float* oc_w       = (const float*)d_in[10];
  const float* oc_b       = (const float*)d_in[11];
  const float* oa_w       = (const float*)d_in[12];
  const float* oa_b       = (const float*)d_in[13];
  const float* op_w       = (const float*)d_in[14];
  const float* op_b       = (const float*)d_in[15];

  char* ws = (char*)d_ws;
  const size_t O_CONV1B = 0;          // bf16 [4][128][512]
  const size_t O_CONV2B = 524288;     // bf16 [4][256][512]
  const size_t O_QKVB   = 1572864;    // bf16 [160][160]
  const size_t O_APB    = 1638400;    // bf16 [128][128]
  const size_t O_OCB    = 1671168;
  const size_t O_OAB    = 1703936;
  const size_t O_OPB    = 1736704;
  const size_t O_H      = 2097152;    // f32  [M][128]
  const size_t O_HXB    = 10485760;   // bf16 [M][128]  (later: elu(conv))
  const size_t O_T1B    = 14680064;   // bf16 [M][128]  (later: elu(attn))
  const size_t O_BIG    = 18874368;   // 16 MiB shared scratch
  const size_t O_ABUF   = 35651584;   // bf16 [M][160]  (later: sb)
  const size_t O_QB     = 40894464;
  const size_t O_KB     = 41418752;
  const size_t O_VTB    = 41943040;   // bf16 [B][128][N]
  const size_t O_AVB    = 46137344;   // bf16 [M][128]
  const size_t O_ATTN   = 50331648;   // f32  [M][128]

  float*          h      = (float*)(ws + O_H);
  unsigned short* hxb    = (unsigned short*)(ws + O_HXB);
  unsigned short* t1b    = (unsigned short*)(ws + O_T1B);
  float*          obuf   = (float*)(ws + O_BIG);
  unsigned short* c1B    = (unsigned short*)(ws + O_CONV1B);
  unsigned short* c2B    = (unsigned short*)(ws + O_CONV2B);
  unsigned short* qkvB   = (unsigned short*)(ws + O_QKVB);
  unsigned short* apB    = (unsigned short*)(ws + O_APB);
  unsigned short* ocB    = (unsigned short*)(ws + O_OCB);
  unsigned short* oaB    = (unsigned short*)(ws + O_OAB);
  unsigned short* opB    = (unsigned short*)(ws + O_OPB);
  unsigned short* abuf   = (unsigned short*)(ws + O_ABUF);
  float*          qkvf   = (float*)(ws + O_BIG);
  unsigned short* qb     = (unsigned short*)(ws + O_QB);
  unsigned short* kb     = (unsigned short*)(ws + O_KB);
  unsigned short* vtb    = (unsigned short*)(ws + O_VTB);
  unsigned short* avb    = (unsigned short*)(ws + O_AVB);
  float*          attnf  = (float*)(ws + O_ATTN);
  unsigned short* cb     = hxb;                         // alias
  unsigned short* ab2    = t1b;                         // alias
  float*          conv2f = (float*)(ws + O_BIG);        // alias (8 MiB)
  float*          attn2f = (float*)(ws + O_BIG + 8388608);
  unsigned short* sb     = abuf;                        // alias

  const int TB = 256;

  transpose_in_kernel<<<cdiv(MPIX * HID, TB), TB, 0, stream>>>(x, h);
  prep_conv_w_kernel<<<cdiv(4 * 128 * 512, TB), TB, 0, stream>>>(conv1_w, c1B, 4 * 128 * 512);
  prep_conv_w_kernel<<<cdiv(4 * 256 * 512, TB), TB, 0, stream>>>(conv2_w, c2B, 4 * 256 * 512);
  prep_lin_w_kernel<<<cdiv(160 * 160, TB), TB, 0, stream>>>(qkv_w, qkvB, 160, 130, 160);
  prep_lin_w_kernel<<<cdiv(128 * 128, TB), TB, 0, stream>>>(ap_w, apB, 128, 128, 128);
  prep_lin_w_kernel<<<cdiv(128 * 128, TB), TB, 0, stream>>>(oc_w, ocB, 128, 128, 128);
  prep_lin_w_kernel<<<cdiv(128 * 128, TB), TB, 0, stream>>>(oa_w, oaB, 128, 128, 128);
  prep_lin_w_kernel<<<cdiv(128 * 128, TB), TB, 0, stream>>>(op_w, opB, 128, 128, 128);

  for (int l = 0; l < 4; ++l) {
    elu_bf16_kernel<<<cdiv(MPIX * HID, TB), TB, 0, stream>>>(h, hxb, MPIX * HID);
    conv_gemm_kernel<<<dim3(256, 1), 128, 0, stream>>>(
        hxb, c1B + (size_t)l * 128 * 512, conv1_b + l * 128,
        (float*)nullptr, t1b, 128, /*act=elu*/1);
    conv_gemm_kernel<<<dim3(256, 2), 128, 0, stream>>>(
        t1b, c2B + (size_t)l * 256 * 512, conv2_b + l * 256,
        obuf, (unsigned short*)nullptr, 256, /*act*/0);
    gate_kernel<<<cdiv(MPIX * HID, TB), TB, 0, stream>>>(h, obuf);
  }

  build_a_kernel<<<cdiv(MPIX * 160, TB), TB, 0, stream>>>(h, pos, abuf);
  gemm1x1_kernel<<<256, 128, 0, stream>>>(abuf, qkvB, qkv_b,
      qkvf, (unsigned short*)nullptr, (float*)nullptr, 160, 10, 160, 0);
  pack_qkv_kernel<<<cdiv(MPIX * 160, TB), TB, 0, stream>>>(qkvf, qb, kb, vtb);

  attn_kernel<<<1024, 32, 0, stream>>>(qb, kb, vtb, avb);

  gemm1x1_kernel<<<256, 128, 0, stream>>>(avb, apB, ap_b,
      attnf, (unsigned short*)nullptr, (float*)nullptr, 128, 8, 128, 0);

  elu_bf16_kernel<<<cdiv(MPIX * HID, TB), TB, 0, stream>>>(h, cb, MPIX * HID);
  elu_bf16_kernel<<<cdiv(MPIX * HID, TB), TB, 0, stream>>>(attnf, ab2, MPIX * HID);
  gemm1x1_kernel<<<256, 128, 0, stream>>>(cb, ocB, oc_b,
      conv2f, (unsigned short*)nullptr, (float*)nullptr, 128, 8, 128, /*elu*/1);
  gemm1x1_kernel<<<256, 128, 0, stream>>>(ab2, oaB, oa_b,
      attn2f, (unsigned short*)nullptr, (float*)nullptr, 128, 8, 128, /*elu*/1);
  add_elu_bf16_kernel<<<cdiv(MPIX * HID, TB), TB, 0, stream>>>(conv2f, attn2f, sb, MPIX * HID);
  gemm1x1_kernel<<<256, 128, 0, stream>>>(sb, opB, op_b,
      (float*)nullptr, (unsigned short*)nullptr, (float*)d_out, 128, 8, 128, /*elu*/1);
}
